// MultiHeadAttention_11476152615204
// MI455X (gfx1250) — matrix operand order repaired and checked by reading, not compiled, not run on hardware
//
#include <hip/hip_runtime.h>
#include <hip/hip_bf16.h>
#include <stdint.h>

// ---------------- problem constants ----------------
#define N_HEADS 16
#define KQ_DIM  64
#define EMB_DIM 1024
#define HDIM    1024          // N_HEADS * KQ_DIM
#define BATCH   2
#define SEQ     2048
#define MTOT    (BATCH*SEQ)   // 4096 token rows

typedef __attribute__((ext_vector_type(16))) _Float16 v16h;
typedef __attribute__((ext_vector_type(8)))  _Float16 v8h;
typedef __attribute__((ext_vector_type(8)))  float    v8f;
typedef __attribute__((ext_vector_type(4)))  float    v4f;

// ---------------- CDNA5 helpers ----------------
static __device__ __forceinline__ void async_g2l_b128(unsigned lds_off,
                                                      unsigned voff,
                                                      const void* sbase) {
  // per-lane: LDS[lds_off] = MEM[sbase + voff], 16 bytes, tracked by ASYNCcnt
  asm volatile("global_load_async_to_lds_b128 %0, %1, %2"
               :: "v"(lds_off), "v"(voff), "s"(sbase) : "memory");
}
static __device__ __forceinline__ void wait_async0() {
  asm volatile("s_wait_asynccnt 0" ::: "memory");
}

static __device__ __forceinline__ v16h frag_cat(v8h lo, v8h hi) {
  v16h r;
#pragma unroll
  for (int i = 0; i < 8; ++i) { r[i] = lo[i]; r[i + 8] = hi[i]; }
  return r;
}
// A-fragment from global: 16 halves at p[0..7] and p[16..23] (K-major row)
static __device__ __forceinline__ v16h load_frag_a(const _Float16* p) {
  const v8h* q = (const v8h*)p;
  return frag_cat(q[0], q[2]);
}
// Packed LDS-resident A-fragment (P): 16 consecutive halves
static __device__ __forceinline__ v16h load_frag16(const _Float16* p) {
  const v8h* q = (const v8h*)p;
  return frag_cat(q[0], q[1]);
}
static __device__ __forceinline__ v8f wmma_f16(v16h a, v16h b, v8f c) {
  return __builtin_amdgcn_wmma_f32_16x16x32_f16(false, a, false, b,
                                                (short)0, c, false, false);
}

// ---------------- kernel 1: f32 -> f16 convert (8 elems/thread) ----------------
__global__ void k_f32_to_f16(const float* __restrict__ in,
                             _Float16* __restrict__ out, int n8) {
  int i = blockIdx.x * blockDim.x + threadIdx.x;   // i indexes groups of 8
  if (i < n8) {
    const v4f* ip = (const v4f*)in + 2 * (size_t)i;
    v4f a = ip[0], b = ip[1];
    v8h o;
#pragma unroll
    for (int j = 0; j < 4; ++j) { o[j] = (_Float16)a[j]; o[j + 4] = (_Float16)b[j]; }
    ((v8h*)out)[i] = o;
  }
}

// ---------------- kernel 2: NT GEMM via WMMA, double-buffered LDS ----------------
// C[M,N] = A[M,K] * Bw[N,K]^T (+bias[N]) (+resid[M,N]) ; all K-contiguous.
// Block: 256 threads = 8 waves; tile 128(M) x 64(N); K-step 64.
__global__ __launch_bounds__(256)
void k_gemm_nt_wmma(const _Float16* __restrict__ A,
                    const _Float16* __restrict__ Bw,
                    float* __restrict__ C,
                    int Mtot, int Ntot, int Ktot,
                    const float* __restrict__ bias,
                    const float* __restrict__ resid) {
  __shared__ __align__(16) _Float16 Bs[2][64 * 72];  // 64 rows x 64 halves (+8 pad)

  const int tid  = threadIdx.x;
  const int wave = tid >> 5;
  const int lane = tid & 31;
  const int half = lane >> 4;
  const int lcol = lane & 15;
  const int m0 = blockIdx.y * 128 + wave * 16;
  const int n0 = blockIdx.x * 64;
  const int r = tid >> 3;      // 0..31 (staging row within half)
  const int c = tid & 7;       // 0..7  (16B chunk within 128B row)

  unsigned bs_base[2];
  bs_base[0] = (unsigned)(size_t)&Bs[0][0];
  bs_base[1] = (unsigned)(size_t)&Bs[1][0];

  v8f acc[4] = {};

  // prologue: stage first tile into buffer 0
  {
    const _Float16* gbase = Bw + (size_t)n0 * Ktot;
    async_g2l_b128(bs_base[0] + r * 144 + c * 16,
                   (unsigned)(r * Ktot * 2 + c * 16), gbase);
    async_g2l_b128(bs_base[0] + (r + 32) * 144 + c * 16,
                   (unsigned)((r + 32) * Ktot * 2 + c * 16), gbase);
  }

  for (int k0 = 0; k0 < Ktot; k0 += 64) {
    const int cur = (k0 >> 6) & 1;
    wait_async0();
    __syncthreads();
    // overlap: kick off next tile's DMA while computing on current buffer
    if (k0 + 64 < Ktot) {
      const int nxt = cur ^ 1;
      const _Float16* gbase = Bw + (size_t)n0 * Ktot + k0 + 64;
      async_g2l_b128(bs_base[nxt] + r * 144 + c * 16,
                     (unsigned)(r * Ktot * 2 + c * 16), gbase);
      async_g2l_b128(bs_base[nxt] + (r + 32) * 144 + c * 16,
                     (unsigned)((r + 32) * Ktot * 2 + c * 16), gbase);
    }
    const _Float16* bs = &Bs[cur][0];

#pragma unroll
    for (int kk = 0; kk < 2; ++kk) {
      const _Float16* ap = A + (size_t)(m0 + lcol) * Ktot + k0 + kk * 32 + half * 8;
      v16h afrag = load_frag_a(ap);
      v16h bf[4];
#pragma unroll
      for (int nt = 0; nt < 4; ++nt)
        bf[nt] = load_frag_a(&bs[(nt * 16 + lcol) * 72 + kk * 32 + half * 8]);
#pragma unroll
      for (int nt = 0; nt < 4; ++nt)
        acc[nt] = wmma_f16(afrag, bf[nt], acc[nt]);
    }
  }

#pragma unroll
  for (int nt = 0; nt < 4; ++nt) {
#pragma unroll
    for (int rr = 0; rr < 8; ++rr) {
      int row = m0 + rr + 8 * half;
      int col = n0 + nt * 16 + lcol;
      float v = acc[nt][rr];
      if (bias)  v += bias[col];
      if (resid) v += resid[(size_t)row * Ntot + col];
      C[(size_t)row * Ntot + col] = v;
    }
  }
}

// ---------------- kernel 3: per-head layernorm + scale + pack f16 ----------------
// X: [MTOT, HDIM] f32 ; Y: [B,H,S,D] f16 ; one wave per (token, head) row of 64.
__global__ __launch_bounds__(256)
void k_ln_head_pack(const float* __restrict__ X,
                    const float* __restrict__ w,
                    const float* __restrict__ bvec,
                    _Float16* __restrict__ Y, float scale) {
  const int wave = threadIdx.x >> 5;
  const int lane = threadIdx.x & 31;
  const int idx = blockIdx.x * 8 + wave;       // 0 .. MTOT*H-1
  const int sg = idx >> 4;                     // token row 0..4095
  const int h  = idx & 15;
  const int b  = sg >> 11;
  const int s  = sg & 2047;

  const float* xp = X + (size_t)sg * HDIM + h * 64;
  float x0 = xp[lane], x1 = xp[lane + 32];

  float sum = x0 + x1;
#pragma unroll
  for (int off = 16; off; off >>= 1) sum += __shfl_xor(sum, off);
  float mu = sum * (1.0f / 64.0f);
  float d0 = x0 - mu, d1 = x1 - mu;
  float vs = d0 * d0 + d1 * d1;
#pragma unroll
  for (int off = 16; off; off >>= 1) vs += __shfl_xor(vs, off);
  float rstd = rsqrtf(vs * (1.0f / 64.0f) + 1e-5f);

  _Float16* yp = Y + (((size_t)b * N_HEADS + h) * SEQ + s) * KQ_DIM;
  yp[lane]      = (_Float16)(((d0 * rstd) * w[lane]      + bvec[lane])      * scale);
  yp[lane + 32] = (_Float16)(((d1 * rstd) * w[lane + 32] + bvec[lane + 32]) * scale);
}

// ---------------- kernel 4: V -> f16 transposed (B,H,D,S) ----------------
__global__ void k_v_pack_t(const float* __restrict__ V, _Float16* __restrict__ VT) {
  int t = blockIdx.x * 256 + threadIdx.x;      // over B*H*D*S
  int s  = t & 2047;
  int d  = (t >> 11) & 63;
  int bh = t >> 17;
  int b = bh >> 4, h = bh & 15;
  VT[t] = (_Float16)V[((size_t)b * SEQ + s) * HDIM + h * 64 + d];
}

// ---------------- kernel 5: flash attention (double-buffered K/V) ----------------
// Grid: (SEQ/128, B*H). Block 256 = 8 waves; wave owns 16 q rows x 64 dims.
__global__ __launch_bounds__(256)
void k_flash_attn(const _Float16* __restrict__ qh,   // [B,H,S,D]
                  const _Float16* __restrict__ kh,   // [B,H,S,D]
                  const _Float16* __restrict__ vth,  // [B,H,D,S]
                  _Float16* __restrict__ aoh) {      // [MTOT, HDIM]
  __shared__ __align__(16) _Float16 Ks[2][64 * 72];       // [j][d]
  __shared__ __align__(16) _Float16 Vs[2][64 * 72];       // [d][j]
  __shared__ __align__(16) _Float16 Ps[8][2][32][16];     // A-frag-ready P per wave

  const int tid  = threadIdx.x;
  const int wave = tid >> 5;
  const int lane = tid & 31;
  const int half = lane >> 4;
  const int lcol = lane & 15;
  const int bh = blockIdx.y;
  const int b = bh >> 4, h = bh & 15;
  const int q0 = blockIdx.x * 128 + wave * 16;
  const int r = tid >> 3;   // staging row (0..31)
  const int c = tid & 7;    // staging 16B chunk

  unsigned ks_base[2], vs_base[2];
  ks_base[0] = (unsigned)(size_t)&Ks[0][0];
  ks_base[1] = (unsigned)(size_t)&Ks[1][0];
  vs_base[0] = (unsigned)(size_t)&Vs[0][0];
  vs_base[1] = (unsigned)(size_t)&Vs[1][0];

  const _Float16* kbase = kh + (size_t)bh * SEQ * KQ_DIM;
  const _Float16* vbase = vth + (size_t)bh * KQ_DIM * SEQ;

  // resident Q fragments (16 rows x 64 dims)
  const _Float16* qbase = qh + (size_t)bh * SEQ * KQ_DIM;
  v16h qfrag[2];
#pragma unroll
  for (int kk = 0; kk < 2; ++kk)
    qfrag[kk] = load_frag_a(qbase + (size_t)(q0 + lcol) * 64 + kk * 32 + half * 8);

  v8f oacc[4] = {};
  float mrow[8], lrow[8];
#pragma unroll
  for (int rr = 0; rr < 8; ++rr) { mrow[rr] = -1e30f; lrow[rr] = 0.0f; }

  // prologue: stage chunk 0 into buffer 0
  {
    const _Float16* kg = kbase;
    const _Float16* vg = vbase;
    async_g2l_b128(ks_base[0] + r * 144 + c * 16,        (unsigned)(r * 128 + c * 16),            kg);
    async_g2l_b128(ks_base[0] + (r + 32) * 144 + c * 16, (unsigned)((r + 32) * 128 + c * 16),     kg);
    async_g2l_b128(vs_base[0] + r * 144 + c * 16,        (unsigned)(r * SEQ * 2 + c * 16),        vg);
    async_g2l_b128(vs_base[0] + (r + 32) * 144 + c * 16, (unsigned)((r + 32) * SEQ * 2 + c * 16), vg);
  }

  for (int j0 = 0; j0 < SEQ; j0 += 64) {
    const int cur = (j0 >> 6) & 1;
    wait_async0();
    __syncthreads();
    if (j0 + 64 < SEQ) {
      const int nxt = cur ^ 1;
      const _Float16* kg = kbase + (size_t)(j0 + 64) * 64;
      const _Float16* vg = vbase + j0 + 64;
      async_g2l_b128(ks_base[nxt] + r * 144 + c * 16,        (unsigned)(r * 128 + c * 16),            kg);
      async_g2l_b128(ks_base[nxt] + (r + 32) * 144 + c * 16, (unsigned)((r + 32) * 128 + c * 16),     kg);
      async_g2l_b128(vs_base[nxt] + r * 144 + c * 16,        (unsigned)(r * SEQ * 2 + c * 16),        vg);
      async_g2l_b128(vs_base[nxt] + (r + 32) * 144 + c * 16, (unsigned)((r + 32) * SEQ * 2 + c * 16), vg);
    }
    const _Float16* ks = &Ks[cur][0];
    const _Float16* vs = &Vs[cur][0];

    // ---- scores S = q . k  (16 x 64), K-dim = 64 ----
    v8f sacc[4] = {};
#pragma unroll
    for (int kk = 0; kk < 2; ++kk) {
      v16h bf[4];
#pragma unroll
      for (int nt = 0; nt < 4; ++nt)
        bf[nt] = load_frag_a(&ks[(nt * 16 + lcol) * 72 + kk * 32 + half * 8]);
#pragma unroll
      for (int nt = 0; nt < 4; ++nt)
        sacc[nt] = wmma_f16(qfrag[kk], bf[nt], sacc[nt]);
    }

    // ---- online softmax (per-lane elements sit on row rr+8*half) ----
#pragma unroll
    for (int rr = 0; rr < 8; ++rr) {
      float mx = fmaxf(fmaxf(sacc[0][rr], sacc[1][rr]),
                       fmaxf(sacc[2][rr], sacc[3][rr]));
#pragma unroll
      for (int off = 1; off < 16; off <<= 1) mx = fmaxf(mx, __shfl_xor(mx, off));
      float mnew = fmaxf(mrow[rr], mx);
      float alpha = __expf(mrow[rr] - mnew);
      mrow[rr] = mnew;
      float rs = 0.0f;
#pragma unroll
      for (int nt = 0; nt < 4; ++nt) {
        float p = __expf(sacc[nt][rr] - mnew);
        sacc[nt][rr] = p;
        rs += p;
      }
#pragma unroll
      for (int off = 1; off < 16; off <<= 1) rs += __shfl_xor(rs, off);
      lrow[rr] = lrow[rr] * alpha + rs;
#pragma unroll
      for (int nt = 0; nt < 4; ++nt) oacc[nt][rr] *= alpha;
    }

    // ---- scatter P into LDS in A-fragment layout (wave-private, no barrier) ----
#pragma unroll
    for (int nt = 0; nt < 4; ++nt) {
#pragma unroll
      for (int rr = 0; rr < 8; ++rr) {
        int row = rr + 8 * half;
        int k = nt * 16 + lcol;
        int kk = k >> 5, kloc = k & 31;
        int lp  = row + (((kloc >> 3) & 1) << 4);
        int idx = (kloc & 7) + ((kloc >> 4) << 3);
        Ps[wave][kk][lp][idx] = (_Float16)sacc[nt][rr];
      }
    }

    // ---- O += P @ V  (K-dim = 64 keys) ----
#pragma unroll
    for (int kk = 0; kk < 2; ++kk) {
      v16h afrag = load_frag16(&Ps[wave][kk][lane][0]);
      v16h bf[4];
#pragma unroll
      for (int nt = 0; nt < 4; ++nt)
        bf[nt] = load_frag_a(&vs[(nt * 16 + lcol) * 72 + kk * 32 + half * 8]);
#pragma unroll
      for (int nt = 0; nt < 4; ++nt)
        oacc[nt] = wmma_f16(afrag, bf[nt], oacc[nt]);
    }
  }

  // ---- epilogue: divide by l, pack to f16 at [token, h*64+d] ----
#pragma unroll
  for (int nt = 0; nt < 4; ++nt) {
#pragma unroll
    for (int rr = 0; rr < 8; ++rr) {
      int row = rr + 8 * half;
      int s = q0 + row;
      size_t off = ((size_t)b * SEQ + s) * HDIM + h * 64 + nt * 16 + lcol;
      aoh[off] = (_Float16)(oacc[nt][rr] / lrow[rr]);
    }
  }
}

// ---------------- host-side orchestration ----------------
extern "C" void kernel_launch(void* const* d_in, const int* in_sizes, int n_in,
                              void* d_out, int out_size, void* d_ws, size_t ws_size,
                              hipStream_t stream) {
  const float* emb  = (const float*)d_in[0];   // [B,S,E]
  const float* Wq   = (const float*)d_in[1];   // [HD,E]
  const float* Wk   = (const float*)d_in[2];
  const float* Wv   = (const float*)d_in[3];
  const float* Wu   = (const float*)d_in[4];   // [E,HD]
  const float* bu   = (const float*)d_in[5];   // [E]
  const float* qn_w = (const float*)d_in[6];
  const float* qn_b = (const float*)d_in[7];
  const float* kn_w = (const float*)d_in[8];
  const float* kn_b = (const float*)d_in[9];
  float* out = (float*)d_out;                  // [B,S,E] f32

  // workspace carve-up (bytes)
  char* ws = (char*)d_ws;
  _Float16* embh = (_Float16*)(ws + 0);              //  8 MB  [MTOT,E]
  _Float16* wqh  = (_Float16*)(ws + 8388608);        //  2 MB  [HD,E]
  _Float16* wkh  = (_Float16*)(ws + 10485760);       //  2 MB
  _Float16* wvh  = (_Float16*)(ws + 12582912);       //  2 MB
  _Float16* wuh  = (_Float16*)(ws + 14680064);       //  2 MB  [E,HD]
  float*    qf   = (float*)   (ws + 16777216);       // 16 MB  [MTOT,HD]
  float*    kf   = (float*)   (ws + 33554432);       // 16 MB
  float*    vf   = (float*)   (ws + 50331648);       // 16 MB
  _Float16* qh   = (_Float16*)(ws + 67108864);       //  8 MB  [B,H,S,D]
  _Float16* kh   = (_Float16*)(ws + 75497472);       //  8 MB
  _Float16* vth  = (_Float16*)(ws + 83886080);       //  8 MB  [B,H,D,S]
  _Float16* aoh  = (_Float16*)(ws + 92274688);       //  8 MB  [MTOT,HD]

  // 1) f32 -> f16 converts (8 elems/thread)
  {
    int n8 = (MTOT * EMB_DIM) / 8;
    k_f32_to_f16<<<(n8 + 255) / 256, 256, 0, stream>>>(emb, embh, n8);
    n8 = (HDIM * EMB_DIM) / 8;
    k_f32_to_f16<<<(n8 + 255) / 256, 256, 0, stream>>>(Wq, wqh, n8);
    k_f32_to_f16<<<(n8 + 255) / 256, 256, 0, stream>>>(Wk, wkh, n8);
    k_f32_to_f16<<<(n8 + 255) / 256, 256, 0, stream>>>(Wv, wvh, n8);
    k_f32_to_f16<<<(n8 + 255) / 256, 256, 0, stream>>>(Wu, wuh, n8);
  }

  // 2) Q/K/V projections: [MTOT,E] x [HD,E]^T -> [MTOT,HD]
  dim3 ggrid(HDIM / 64, MTOT / 128);
  k_gemm_nt_wmma<<<ggrid, 256, 0, stream>>>(embh, wqh, qf, MTOT, HDIM, EMB_DIM, nullptr, nullptr);
  k_gemm_nt_wmma<<<ggrid, 256, 0, stream>>>(embh, wkh, kf, MTOT, HDIM, EMB_DIM, nullptr, nullptr);
  k_gemm_nt_wmma<<<ggrid, 256, 0, stream>>>(embh, wvh, vf, MTOT, HDIM, EMB_DIM, nullptr, nullptr);

  // 3) per-head layernorm; fold 1/sqrt(D)=0.125 into q
  k_ln_head_pack<<<(MTOT * N_HEADS) / 8, 256, 0, stream>>>(qf, qn_w, qn_b, qh, 0.125f);
  k_ln_head_pack<<<(MTOT * N_HEADS) / 8, 256, 0, stream>>>(kf, kn_w, kn_b, kh, 1.0f);

  // 4) pack V transposed
  k_v_pack_t<<<(BATCH * N_HEADS * KQ_DIM * SEQ) / 256, 256, 0, stream>>>(vf, vth);

  // 5) flash attention
  dim3 fgrid(SEQ / 128, BATCH * N_HEADS);
  k_flash_attn<<<fgrid, 256, 0, stream>>>(qh, kh, vth, aoh);

  // 6) output projection + bias + residual: [MTOT,HD] x [E,HD]^T + bu + emb
  dim3 ogrid(EMB_DIM / 64, MTOT / 128);
  k_gemm_nt_wmma<<<ogrid, 256, 0, stream>>>(aoh, wuh, out, MTOT, EMB_DIM, HDIM, bu, emb);
}
